// GPRGNN_48163763258021
// MI455X (gfx1250) — matrix operand order repaired and checked by reading, not compiled, not run on hardware
//
#include <hip/hip_runtime.h>
#include <hip/hip_fp16.h>

// ---------------------------------------------------------------------------
// GPRGNN on MI455X (gfx1250):
//   - x pre-packed to f16 WMMA A-fragment layout (one streaming pass)
//   - fused WMMA MLP: W1/W2 f16-packed in LDS (persistent blocks);
//     per-strip A staged via global_load_async_to_lds_b128 (ASYNCcnt)
//   - gcn_norm via histogram + scan -> CSR with precomputed edge coefs
//   - 10 propagation steps as unrolled row-gather kernels (no float atomics)
// ---------------------------------------------------------------------------

typedef __attribute__((ext_vector_type(16))) _Float16 v16h;
typedef __attribute__((ext_vector_type(8)))  float    v8f;

union CFrag { v8f v; float e[8]; };

#define IN_DIM   512
#define HID_DIM  256
#define OUT_DIM  64

// LDS map (in halves)
#define LDS_W1_HALFS (IN_DIM * HID_DIM)          // 131072 -> 256 KB
#define LDS_W2_HALFS (HID_DIM * OUT_DIM)         // 16384  ->  32 KB
#define LDS_H1_HALFS (8 * 512)                   // 4096   ->   8 KB
#define LDS_A_HALFS  (16 * 32 * 16)              // 8192   ->  16 KB (one strip)
#define OFF_W2 (LDS_W1_HALFS)
#define OFF_H1 (OFF_W2 + LDS_W2_HALFS)
#define OFF_A  (OFF_H1 + LDS_H1_HALFS)
#define LDS_TOTAL_BYTES ((OFF_A + LDS_A_HALFS) * 2)   // 319488 B = 312 KB

// ============ x -> f16 A-fragment pack (streaming, coalesced writes) =======
// layout: xh[((strip*16 + kt)*32 + lane)*16 + j]
//   row = strip*16 + (lane&15)
//   K   = kt*32 + (lane>>4)*8 + (j<8 ? j : 16 + (j-8))
__global__ void __launch_bounds__(256)
xpack_kernel(const float* __restrict__ x, _Float16* __restrict__ xh,
             int nItems) {
  int g = blockIdx.x * blockDim.x + threadIdx.x;
  if (g >= nItems) return;                  // nItems = nStrips*1024
  const int jh    = g & 1;
  const int lane  = (g >> 1) & 31;
  const int kt    = (g >> 6) & 15;
  const int strip = g >> 10;
  const int row = strip * 16 + (lane & 15);
  const int Kb  = kt * 32 + ((lane >> 4) << 3) + (jh << 4);
  const float4* p = (const float4*)(x + (size_t)row * IN_DIM + Kb);
  const float4 a = p[0];
  const float4 b = p[1];
  union { _Float16 h[8]; uint4 u; } o;
  o.h[0] = (_Float16)a.x; o.h[1] = (_Float16)a.y;
  o.h[2] = (_Float16)a.z; o.h[3] = (_Float16)a.w;
  o.h[4] = (_Float16)b.x; o.h[5] = (_Float16)b.y;
  o.h[6] = (_Float16)b.z; o.h[7] = (_Float16)b.w;
  *(uint4*)(xh + ((size_t)((strip * 16 + kt) * 32 + lane) * 16 + jh * 8)) = o.u;
}

// ======================= fused MLP (WMMA) ==================================
__global__ void __launch_bounds__(256)
mlp_wmma_kernel(const _Float16* __restrict__ xh,
                const float* __restrict__ W1,
                const float* __restrict__ b1,
                const float* __restrict__ W2,
                const float* __restrict__ b2,
                float* __restrict__ hout,
                int nStrips) {
  extern __shared__ _Float16 smem[];
  _Float16* w1s = smem;
  _Float16* w2s = smem + OFF_W2;
  _Float16* h1s = smem + OFF_H1;
  _Float16* aS  = smem + OFF_A;

  const int tid  = threadIdx.x;
  const int lane = tid & 31;
  const int w    = tid >> 5;

  // pack W1/W2 (f32 -> f16) into B-fragment layout:
  // [kt][nt][lane][j], K = kt*32 + (lane>>4)*8 + (j<8 ? j : 16 + (j-8)), N = nt*16 + (lane&15)
  for (int i = tid; i < LDS_W1_HALFS; i += 256) {
    int j = i & 15, ln = (i >> 4) & 31, nt = (i >> 9) & 15, kt = i >> 13;
    int K  = kt * 32 + ((ln >> 4) << 3) + (j & 7) + ((j >> 3) << 4);
    int Nc = nt * 16 + (ln & 15);
    w1s[i] = (_Float16)W1[K * HID_DIM + Nc];
  }
  for (int i = tid; i < LDS_W2_HALFS; i += 256) {
    int j = i & 15, ln = (i >> 4) & 31, nt = (i >> 9) & 3, kt = i >> 11;
    int K  = kt * 32 + ((ln >> 4) << 3) + (j & 7) + ((j >> 3) << 4);
    int Nc = nt * 16 + (ln & 15);
    w2s[i] = (_Float16)W2[K * OUT_DIM + Nc];
  }
  __syncthreads();

  for (int strip = blockIdx.x; strip < nStrips; strip += gridDim.x) {
    const int mBase = strip << 4;

    // ---- async-stage this strip's packed A (16 KB) into LDS --------------
    // 8 waves x 4 instrs x 32 lanes x 16 B = 16 KB, tracked by ASYNCcnt.
    {
      const _Float16* gbase = xh + (size_t)strip * LDS_A_HALFS;
#pragma unroll
      for (int q = 0; q < 4; ++q) {
        const unsigned idx = (unsigned)((w * 4 + q) * 32 + lane);   // 0..1023
        const unsigned ldsOff = (unsigned)(OFF_A * 2) + idx * 16;   // bytes
        const unsigned long long gaddr =
            (unsigned long long)(uintptr_t)(gbase + (size_t)idx * 8);
        asm volatile("global_load_async_to_lds_b128 %0, %1, off"
                     :: "v"(ldsOff), "v"(gaddr)
                     : "memory");
      }
      asm volatile("s_wait_asynccnt 0x0" ::: "memory");
    }
    __syncthreads();

    // ---- GEMM1: 16x256 strip, 2 N-tiles per wave, K=512 (16 chunks) ------
    v8f acc0 = {};
    v8f acc1 = {};
#pragma unroll
    for (int kt = 0; kt < 16; ++kt) {
      v16h a      = *(const v16h*)(aS + (((kt << 5) + lane) << 4));
      v16h bfrag0 = *(const v16h*)(w1s + (((kt * 16 + w * 2 + 0) << 9) + lane * 16));
      v16h bfrag1 = *(const v16h*)(w1s + (((kt * 16 + w * 2 + 1) << 9) + lane * 16));
      acc0 = __builtin_amdgcn_wmma_f32_16x16x32_f16(false, a, false, bfrag0,
                                                    (short)0, acc0, false, false);
      acc1 = __builtin_amdgcn_wmma_f32_16x16x32_f16(false, a, false, bfrag1,
                                                    (short)0, acc1, false, false);
    }

    // ---- bias + relu + stage h1 strip to LDS in A-fragment order ---------
#pragma unroll
    for (int t = 0; t < 2; ++t) {
      CFrag c; c.v = t ? acc1 : acc0;
      const int cBase = (w * 2 + t) * 16;
      const int col   = cBase + (lane & 15);
      const float bias = b1[col];
      const int kt2 = col >> 5;
      const int kk  = col & 31;
      const int j   = ((kk & 16) >> 1) + (kk & 7);
      const int lhi = ((kk >> 3) & 1) << 4;
#pragma unroll
      for (int r = 0; r < 8; ++r) {
        const int M = r + ((lane >> 4) << 3);
        float v = c.e[r] + bias;
        v = v > 0.0f ? v : 0.0f;
        h1s[(kt2 << 9) + ((M + lhi) << 4) + j] = (_Float16)v;
      }
    }
    __syncthreads();

    // ---- GEMM2: 16x64 out, waves 0..3, K=256 (8 chunks) ------------------
    if (w < 4) {
      v8f acc2 = {};
#pragma unroll
      for (int kt2 = 0; kt2 < 8; ++kt2) {
        v16h a2 = *(const v16h*)(h1s + (kt2 << 9) + lane * 16);
        v16h bf = *(const v16h*)(w2s + (((kt2 * 4 + w) << 9) + lane * 16));
        acc2 = __builtin_amdgcn_wmma_f32_16x16x32_f16(false, a2, false, bf,
                                                      (short)0, acc2, false, false);
      }
      CFrag c2; c2.v = acc2;
      const int col = w * 16 + (lane & 15);
      const float bias = b2[col];
#pragma unroll
      for (int r = 0; r < 8; ++r) {
        const int grow = mBase + r + ((lane >> 4) << 3);
        hout[(size_t)grow * OUT_DIM + col] = c2.e[r] + bias;
      }
    }
    __syncthreads();
  }
}

// ======================= CSR build =========================================
__global__ void zero_int_kernel(int* __restrict__ p, int n) {
  int i = blockIdx.x * blockDim.x + threadIdx.x;
  if (i < n) p[i] = 0;
}

__global__ void hist_kernel(const int* __restrict__ ei, int E,
                            int* __restrict__ cnt) {
  int i = blockIdx.x * blockDim.x + threadIdx.x;
  int stride = gridDim.x * blockDim.x;
  for (int e = i; e < E; e += stride)
    atomicAdd(&cnt[ei[E + e]], 1);
}

__global__ void dis_kernel(const int* __restrict__ cnt,
                           float* __restrict__ dis, int n) {
  int i = blockIdx.x * blockDim.x + threadIdx.x;
  if (i < n) dis[i] = rsqrtf((float)cnt[i] + 1.0f);
}

__global__ void __launch_bounds__(256)
scan1_kernel(const int* __restrict__ cnt, int* __restrict__ rowptr,
             int* __restrict__ partial, int n) {
  __shared__ int sdata[256];
  int t = threadIdx.x;
  int i = blockIdx.x * 256 + t;
  int v = (i < n) ? cnt[i] : 0;
  sdata[t] = v;
  __syncthreads();
#pragma unroll
  for (int off = 1; off < 256; off <<= 1) {
    int xval = (t >= off) ? sdata[t - off] : 0;
    __syncthreads();
    sdata[t] += xval;
    __syncthreads();
  }
  if (i < n) rowptr[i] = sdata[t] - v;   // exclusive
  if (t == 255) partial[blockIdx.x] = sdata[255];
}

__global__ void scan2_kernel(int* __restrict__ partial, int nb) {
  if (blockIdx.x == 0 && threadIdx.x == 0) {
    int acc = 0;
    for (int i = 0; i < nb; ++i) { int v = partial[i]; partial[i] = acc; acc += v; }
  }
}

__global__ void scan3_kernel(int* __restrict__ rowptr,
                             const int* __restrict__ partial,
                             int* __restrict__ cursor, int n) {
  int i = blockIdx.x * blockDim.x + threadIdx.x;
  if (i < n) {
    int v = rowptr[i] + partial[i >> 8];
    rowptr[i] = v;
    cursor[i] = v;
  }
}

__global__ void fill_kernel(const int* __restrict__ ei, int E,
                            const float* __restrict__ dis,
                            int* __restrict__ cursor,
                            int2* __restrict__ meta) {
  int i = blockIdx.x * blockDim.x + threadIdx.x;
  int stride = gridDim.x * blockDim.x;
  for (int e = i; e < E; e += stride) {
    int s = ei[e];
    int d = ei[E + e];
    int pos = atomicAdd(&cursor[d], 1);
    meta[pos] = make_int2(s, __float_as_int(dis[s] * dis[d]));
  }
}

// ======================= propagation (gather, fused fold) ==================
// One wave per destination row; lane owns feature cols {2*lane, 2*lane+1}.
// Unrolled x4 with two accumulators -> 4 concurrent L2 gathers in flight.
__global__ void __launch_bounds__(256)
prop_row_kernel(const int* __restrict__ rowptr,
                const int2* __restrict__ meta,
                const float* __restrict__ dis,
                const float* __restrict__ cur,
                float* __restrict__ nxt,
                float* __restrict__ out,
                const float* __restrict__ temp,
                int k, int n, int E) {
  int wv   = blockIdx.x * (blockDim.x >> 5) + (threadIdx.x >> 5);
  int lane = threadIdx.x & 31;
  int nw   = gridDim.x * (blockDim.x >> 5);
  for (int d0 = wv; d0 < n; d0 += nw) {
    const int d = __builtin_amdgcn_readfirstlane(d0);   // SGPR row id
    const int beg = rowptr[d];
    const int end = (d + 1 < n) ? rowptr[d + 1] : E;
    const float dd = dis[d];

    const float2 self = ((const float2*)(cur + (size_t)d * 64))[lane];
    float2 a0, a1;
    a0.x = dd * dd * self.x;
    a0.y = dd * dd * self.y;
    a1.x = 0.0f;
    a1.y = 0.0f;

    int p = beg;
    for (; p + 3 < end; p += 4) {
      const int2 m0 = meta[p + 0];
      const int2 m1 = meta[p + 1];
      const int2 m2 = meta[p + 2];
      const int2 m3 = meta[p + 3];
      // lookahead: pull next group's first source row toward the WGP
      {
        const int pn = (p + 4 < end) ? (p + 4) : (end - 1);
        const int sn = meta[pn].x;
        __builtin_prefetch(cur + (size_t)sn * 64, 0, 0);
      }
      const float2 v0 = ((const float2*)(cur + (size_t)m0.x * 64))[lane];
      const float2 v1 = ((const float2*)(cur + (size_t)m1.x * 64))[lane];
      const float2 v2 = ((const float2*)(cur + (size_t)m2.x * 64))[lane];
      const float2 v3 = ((const float2*)(cur + (size_t)m3.x * 64))[lane];
      const float c0 = __int_as_float(m0.y), c1 = __int_as_float(m1.y);
      const float c2 = __int_as_float(m2.y), c3 = __int_as_float(m3.y);
      a0.x += c0 * v0.x; a0.y += c0 * v0.y;
      a1.x += c1 * v1.x; a1.y += c1 * v1.y;
      a0.x += c2 * v2.x; a0.y += c2 * v2.y;
      a1.x += c3 * v3.x; a1.y += c3 * v3.y;
    }
    for (; p < end; ++p) {
      const int2 m = meta[p];
      const float c = __int_as_float(m.y);
      const float2 v = ((const float2*)(cur + (size_t)m.x * 64))[lane];
      a0.x += c * v.x; a0.y += c * v.y;
    }

    float2 acc;
    acc.x = a0.x + a1.x;
    acc.y = a0.y + a1.y;

    ((float2*)(nxt + (size_t)d * 64))[lane] = acc;

    const float g = temp[k];
    float2* od = (float2*)(out + (size_t)d * 64);
    float2 o;
    if (k == 1) {
      const float t0 = temp[0];
      o.x = t0 * self.x + g * acc.x;
      o.y = t0 * self.y + g * acc.y;
    } else {
      o = od[lane];
      o.x += g * acc.x;
      o.y += g * acc.y;
    }
    od[lane] = o;
  }
}

// ======================= host launch =======================================
extern "C" void kernel_launch(void* const* d_in, const int* in_sizes, int n_in,
                              void* d_out, int out_size, void* d_ws, size_t ws_size,
                              hipStream_t stream) {
  const float* x    = (const float*)d_in[0];
  const int*   ei   = (const int*)d_in[1];
  const float* W1   = (const float*)d_in[2];
  const float* b1   = (const float*)d_in[3];
  const float* W2   = (const float*)d_in[4];
  const float* b2   = (const float*)d_in[5];
  const float* temp = (const float*)d_in[6];

  const int N = in_sizes[0] / IN_DIM;      // 100000
  const int E = in_sizes[1] / 2;           // 3200000
  const int total = N * OUT_DIM;           // 6.4M
  float* out = (float*)d_out;

  // workspace layout
  float*    h      = (float*)d_ws;                  // N*64 f32
  float*    buf0   = h    + (size_t)total;          // N*64 f32
  float*    buf1   = buf0 + (size_t)total;          // N*64 f32
  float*    dis    = buf1 + (size_t)total;          // N f32
  int*      cnt    = (int*)(dis + N);               // N int
  int*      rowptr = cnt + N;                       // N int
  int*      cursor = rowptr + N;                    // N int
  int*      partial= cursor + N;                    // <=1024 int
  int2*     meta   = (int2*)(partial + 1024);       // E int2
  _Float16* xh     = (_Float16*)(meta + E);         // N*512 f16 (packed A)

  const int nStrips    = N / 16;                    // 6250
  const int nBlocks    = (N + 255) / 256;           // 391
  const int edgeBlocks = 2048;

  // 0) pack x into f16 A-fragment layout
  const int nItems = nStrips * 1024;
  xpack_kernel<<<(nItems + 255) / 256, 256, 0, stream>>>(x, xh, nItems);

  // 1) MLP (WMMA + async LDS staging) -> h
  mlp_wmma_kernel<<<1024, 256, LDS_TOTAL_BYTES, stream>>>(xh, W1, b1, W2, b2, h,
                                                          nStrips);

  // 2) degree histogram + normalization
  zero_int_kernel<<<nBlocks, 256, 0, stream>>>(cnt, N);
  hist_kernel<<<edgeBlocks, 256, 0, stream>>>(ei, E, cnt);
  dis_kernel<<<nBlocks, 256, 0, stream>>>(cnt, dis, N);

  // 3) exclusive scan -> rowptr, cursor
  scan1_kernel<<<nBlocks, 256, 0, stream>>>(cnt, rowptr, partial, N);
  scan2_kernel<<<1, 32, 0, stream>>>(partial, nBlocks);
  scan3_kernel<<<nBlocks, 256, 0, stream>>>(rowptr, partial, cursor, N);

  // 4) CSR fill with precomputed coefficients
  fill_kernel<<<edgeBlocks, 256, 0, stream>>>(ei, E, dis, cursor, meta);

  // 5) 10 propagation steps (gather only; fold fused). cur starts at h.
  float* cur = h;
  float* nxt = buf0;
  float* spare = buf1;
  for (int k = 1; k <= 10; ++k) {
    prop_row_kernel<<<2048, 256, 0, stream>>>(rowptr, meta, dis, cur, nxt, out,
                                              temp, k, N, E);
    float* old_cur = cur;
    cur = nxt;
    nxt = (old_cur == h) ? spare : old_cur;
  }
}
